// TemplateDeformNet_45938970198403
// MI455X (gfx1250) — matrix operand order repaired and checked by reading, not compiled, not run on hardware
//
#include <hip/hip_runtime.h>
#include <hip/hip_bf16.h>
#include <math.h>

// ---------------- problem constants (from reference) ----------------
#define BB   4
#define TT   8192
#define SS   4096
#define DD   256
#define GG   512
#define HH   256
#define KK   8
#define INP  771          // 3 + D + G
#define KSM  259          // 3 + D  (template + local part of ni)
#define NIP  288          // KSM padded to 9 k-tiles of 32
#define BT   (BB*TT)      // 32768 rows
#define NC1  640          // fused GEMM1 columns: 256 (W1) | 256 (Ws) | 128 (M1)
#define MLOOP 4           // M-tiles per wave in the GEMM kernels
#define DISP_SCALE 0.3f

// sched_group_barrier masks
#define SGB_WMMA   0x008
#define SGB_VMEMRD 0x020
#define SGB_DSRD   0x100

typedef __attribute__((ext_vector_type(16))) __bf16 v16bf;
typedef __attribute__((ext_vector_type(8)))  __bf16 v8bf;
typedef __attribute__((ext_vector_type(8)))  float  v8f;

union V16U { v16bf v; v8bf h[2]; };

static __device__ __forceinline__ unsigned short f2bf(float f) {
    unsigned int u = __float_as_uint(f);
    unsigned int lsb = (u >> 16) & 1u;
    u += 0x7FFFu + lsb;                 // round-to-nearest-even
    return (unsigned short)(u >> 16);
}

// CDNA5 async global->LDS copy (ASYNCcnt tracked, no VGPR data round-trip).
static __device__ __forceinline__ void async_copy16(unsigned lds_off, const void* gaddr) {
    asm volatile("global_load_async_to_lds_b128 %0, %1, off"
                 :: "v"(lds_off), "v"(gaddr) : "memory");
}
static __device__ __forceinline__ void wait_async0() {
    asm volatile("s_wait_asynccnt 0" ::: "memory");
}

// ---------------- prep: WcatT [640 x 288] bf16 (N-major, zero-padded K) ----------
__global__ void prep_wcat_kernel(const float* __restrict__ W1, const float* __restrict__ Ws,
                                 const float* __restrict__ M1, unsigned short* __restrict__ wT) {
    int i = blockIdx.x * 256 + threadIdx.x;
    if (i >= NC1 * NIP) return;
    int n = i / NIP, k = i % NIP;
    float v = 0.f;
    if (k < KSM) {
        if (n < 256)      v = W1[k * 256 + n];
        else if (n < 512) v = Ws[k * 256 + (n - 256)];
        else              v = M1[k * 128 + (n - 512)];
    }
    wT[i] = f2bf(v);
}

// ---------------- prep: W2T [256 x 256], M2T [64 x 128] bf16 (N-major) ----------
__global__ void prep_w2m2_kernel(const float* __restrict__ W2, const float* __restrict__ M2,
                                 unsigned short* __restrict__ w2T, unsigned short* __restrict__ m2T) {
    int i = blockIdx.x * 256 + threadIdx.x;
    if (i < 256 * 256) {
        int n = i / 256, k = i % 256;
        w2T[i] = f2bf(W2[k * 256 + n]);
    } else if (i < 256 * 256 + 64 * 128) {
        int j = i - 256 * 256;
        int n = j / 128, k = j % 128;
        m2T[j] = f2bf(M2[k * 64 + n]);
    }
}

// -------- prep: per-batch constant contribution of global_feat + biases --------
__global__ void prep_gbias_kernel(const float* __restrict__ gf,
                                  const float* __restrict__ W1, const float* __restrict__ b1,
                                  const float* __restrict__ Ws, const float* __restrict__ bs,
                                  const float* __restrict__ M1, const float* __restrict__ mb1,
                                  float* __restrict__ gbias) {
    int i = blockIdx.x * 256 + threadIdx.x;
    if (i >= BB * NC1) return;
    int b = i / NC1, n = i % NC1;
    float acc;
    if (n < 256)      acc = b1[n];
    else if (n < 512) acc = bs[n - 256];
    else              acc = mb1[n - 512];
    for (int g = 0; g < GG; ++g) {
        float x = gf[b * GG + g];
        int k = KSM + g;
        float w = (n < 256) ? W1[k * 256 + n]
                : (n < 512) ? Ws[k * 256 + (n - 256)]
                            : M1[k * 128 + (n - 512)];
        acc = fmaf(x, w, acc);
    }
    gbias[i] = acc;
}

// ---------------- stage 1: distances + top-8 + weighted gather → ni (bf16) ------
__global__ void local_feat_kernel(const float* __restrict__ tmpl,
                                  const float* __restrict__ surf,
                                  const float* __restrict__ pf,
                                  unsigned short* __restrict__ ni) {
    __shared__ __align__(16) float sxyz[SS * 3];
    const int tid = threadIdx.x, lane = tid & 31, wave = tid >> 5;
    const int pt = blockIdx.x * 8 + wave;       // 8 points per block, one batch per block
    const int b  = pt >> 13;                    // / TT

    // async-stage this batch's surf coords (48KB) into LDS
    {
        const float* src = surf + (size_t)b * SS * 3;
        for (int i = tid; i < (SS * 3) / 4; i += 256)
            async_copy16((unsigned)(uintptr_t)&sxyz[i * 4], src + i * 4);
        wait_async0();
    }
    __syncthreads();

    const float tx = tmpl[(size_t)pt * 3 + 0];
    const float ty = tmpl[(size_t)pt * 3 + 1];
    const float tz = tmpl[(size_t)pt * 3 + 2];

    float d[KK]; int id[KK];
#pragma unroll
    for (int j = 0; j < KK; ++j) { d[j] = 3.4e38f; id[j] = -1; }

    // per-lane top-8 over a strided slice of S
    for (int i = lane; i < SS; i += 32) {
        float dx = tx - sxyz[3 * i], dy = ty - sxyz[3 * i + 1], dz = tz - sxyz[3 * i + 2];
        float c = dx * dx + dy * dy + dz * dz;
        if (c < d[KK - 1]) {
            d[KK - 1] = c; id[KK - 1] = i;
#pragma unroll
            for (int j = KK - 1; j > 0; --j) {
                if (d[j] < d[j - 1]) {
                    float td = d[j]; d[j] = d[j - 1]; d[j - 1] = td;
                    int   ti = id[j]; id[j] = id[j - 1]; id[j - 1] = ti;
                }
            }
        }
    }

    // wave-wide merge: 8 rounds of butterfly argmin over each lane's head
    float gd[KK]; int gi[KK]; int p = 0;
#pragma unroll
    for (int k = 0; k < KK; ++k) {
        float v  = (p < KK) ? d[p] : 3.4e38f;
        int   vi = (p < KK) ? id[p] : -1;
        for (int off = 16; off > 0; off >>= 1) {
            float ov  = __shfl_xor(v, off, 32);
            int   ovi = __shfl_xor(vi, off, 32);
            if (ov < v) { v = ov; vi = ovi; }
        }
        gd[k] = v; gi[k] = vi;
        if (p < KK && id[p] == vi) p++;          // indices unique → exact match
    }

    // inverse-distance weights (uniform across the wave)
    float wk[KK], wsum = 0.f;
#pragma unroll
    for (int k = 0; k < KK; ++k) { wk[k] = 1.f / (sqrtf(fmaxf(gd[k], 0.f)) + 1e-8f); wsum += wk[k]; }
    float inv = 1.f / wsum;

    // gather + weighted sum: lane handles dims lane, lane+32, ... (8 dims)
    float acc[8];
#pragma unroll
    for (int j = 0; j < 8; ++j) acc[j] = 0.f;
#pragma unroll
    for (int k = 0; k < KK; ++k) {
        const float* prow = pf + ((size_t)b * SS + gi[k]) * DD;
        float w = wk[k] * inv;
#pragma unroll
        for (int j = 0; j < 8; ++j) acc[j] = fmaf(w, prow[lane + 32 * j], acc[j]);
    }

    // write bf16 ni row: [tx,ty,tz | local(256) | zero-pad(29)]
    size_t base = (size_t)pt * NIP;
    if (lane < 3) ni[base + lane] = f2bf(lane == 0 ? tx : (lane == 1 ? ty : tz));
#pragma unroll
    for (int j = 0; j < 8; ++j) ni[base + 3 + lane + 32 * j] = f2bf(acc[j]);
    if (lane >= 3) ni[base + 256 + lane] = 0;    // cols 259..287
}

// ---------------- stage 2: fused GEMM1 (ni[BT,288] @ WcatT → 640 cols) ----------
// B register-resident; A tiles async double-buffered in LDS; DS-read/WMMA
// interleave pinned with sched_group_barrier (2 load-pairs ahead of each WMMA).
__global__ void gemm1_kernel(const unsigned short* __restrict__ ni,
                             const unsigned short* __restrict__ wT,     // [640,288] N-major
                             const float* __restrict__ gbias,           // [4,640]
                             unsigned short* __restrict__ h1,           // [BT,256] bf16
                             float* __restrict__ skip,                  // [BT,256] f32
                             unsigned short* __restrict__ m1) {         // [BT,128] bf16
    __shared__ __align__(16) unsigned short sA[2][16 * NIP];
    const int tid = threadIdx.x, lane = tid & 31, wave = tid >> 5;
    const int mbase = blockIdx.x * MLOOP;
    const int ntile = blockIdx.y * 8 + wave;     // 0..39
    const int nlo = lane & 15, hi = lane >> 4;
    const int ncol = ntile * 16 + nlo;

    // B fragments: loaded once, reused for all MLOOP M-tiles (72 VGPRs)
    V16U bfr[NIP / 32];
    {
        const unsigned short* bptr = wT + (size_t)ncol * NIP;
#pragma unroll
        for (int kt = 0; kt < NIP / 32; ++kt) {
            int kb = kt * 32 + hi * 8;           // B: lanes 0-15 K0..7,16..23; 16-31 K8..15,24..31
            bfr[kt].h[0] = *(const v8bf*)(bptr + kb);
            bfr[kt].h[1] = *(const v8bf*)(bptr + kb + 16);
        }
    }

    // prologue: async-stage first A tile (16 contiguous rows = 9KB)
    {
        const unsigned short* src = ni + (size_t)mbase * 16 * NIP;
        for (int i = tid; i < (16 * NIP) / 8; i += 256)
            async_copy16((unsigned)(uintptr_t)&sA[0][i * 8], src + i * 8);
        wait_async0();
    }
    __syncthreads();

    for (int mi = 0; mi < MLOOP; ++mi) {
        const int mtile = mbase + mi;
        if (mi + 1 < MLOOP) {                    // prefetch next A tile into other buffer
            const unsigned short* src = ni + (size_t)(mtile + 1) * 16 * NIP;
            unsigned short* dstb = &sA[(mi + 1) & 1][0];
            for (int i = tid; i < (16 * NIP) / 8; i += 256)
                async_copy16((unsigned)(uintptr_t)(dstb + i * 8), src + i * 8);
        }

        const unsigned short* aB = &sA[mi & 1][0] + nlo * NIP;
        v8f c = {};
        {
            V16U afr[2];                          // 2-deep rotating A buffer (16 VGPRs)
            afr[0].h[0] = *(const v8bf*)(aB + hi * 8);
            afr[0].h[1] = *(const v8bf*)(aB + hi * 8 + 16);
#pragma unroll
            for (int kt = 0; kt < NIP / 32; ++kt) {
                if (kt + 1 < NIP / 32) {
                    int ka = (kt + 1) * 32 + hi * 8;
                    afr[(kt + 1) & 1].h[0] = *(const v8bf*)(aB + ka);
                    afr[(kt + 1) & 1].h[1] = *(const v8bf*)(aB + ka + 16);
                }
                c = __builtin_amdgcn_wmma_f32_16x16x32_bf16(false, afr[kt & 1].v, false, bfr[kt].v, (short)0, c, false, false);
            }
            // pin schedule: [DS x4] then 7x[WMMA, DS x2] then [WMMA x2]
            __builtin_amdgcn_sched_group_barrier(SGB_DSRD, 4, 0);
#pragma unroll
            for (int g = 0; g < 7; ++g) {
                __builtin_amdgcn_sched_group_barrier(SGB_WMMA, 1, 0);
                __builtin_amdgcn_sched_group_barrier(SGB_DSRD, 2, 0);
            }
            __builtin_amdgcn_sched_group_barrier(SGB_WMMA, 2, 0);
        }

        const int rbase = mtile * 16 + hi * 8;
        const int batch = mtile / (TT / 16);
        const float gb = gbias[batch * NC1 + ncol];
#pragma unroll
        for (int r = 0; r < 8; ++r) {
            int row = rbase + r;
            float v = c[r] + gb;
            if (ncol < 256) {
                h1[(size_t)row * 256 + ncol] = f2bf(v > 0.f ? v : 0.f);
            } else if (ncol < 512) {
                skip[(size_t)row * 256 + (ncol - 256)] = v;       // ni@Ws + bs (no relu)
            } else {
                m1[(size_t)row * 128 + (ncol - 512)] = f2bf(v > 0.f ? v : 0.f);
            }
        }
        if (mi + 1 < MLOOP) { wait_async0(); __syncthreads(); }
    }
}

// ---------------- stage 3: fused GEMM2 (h1@W2 +skip → h2 ; m1@M2 → m2) ----------
// B register-resident; A via rotating 2-deep global loads; VMEM/WMMA interleave pinned.
__global__ void gemm2_kernel(const unsigned short* __restrict__ h1,
                             const unsigned short* __restrict__ w2T,   // [256,256]
                             const float* __restrict__ b2,
                             const float* __restrict__ skip,
                             const unsigned short* __restrict__ m1,
                             const unsigned short* __restrict__ m2T,   // [64,128]
                             const float* __restrict__ mb2,
                             float* __restrict__ h2,                   // [BT,256]
                             float* __restrict__ m2o) {                // [BT,64]
    const int gw = blockIdx.x * 8 + (threadIdx.x >> 5);  // (BT/16/MLOOP)*20 waves
    const int lane = threadIdx.x & 31;
    const int ntile = gw % 20;
    const int mbase = (gw / 20) * MLOOP;
    const int nlo = lane & 15, hi = lane >> 4;

    if (ntile < 16) {                                    // h-branch, K=256
        const int ncol = ntile * 16 + nlo;
        V16U bfr[8];
        const unsigned short* bptr = w2T + (size_t)ncol * 256;
#pragma unroll
        for (int kt = 0; kt < 8; ++kt) {
            int kb = kt * 32 + hi * 8;
            bfr[kt].h[0] = *(const v8bf*)(bptr + kb);
            bfr[kt].h[1] = *(const v8bf*)(bptr + kb + 16);
        }
        const float bias = b2[ncol];
        for (int mi = 0; mi < MLOOP; ++mi) {
            const int mtile = mbase + mi;
            const unsigned short* ap = h1 + (size_t)(mtile * 16 + nlo) * 256;
            v8f c = {};
            {
                V16U afr[2];
                afr[0].h[0] = *(const v8bf*)(ap + hi * 8);
                afr[0].h[1] = *(const v8bf*)(ap + hi * 8 + 16);
#pragma unroll
                for (int kt = 0; kt < 8; ++kt) {
                    if (kt + 1 < 8) {
                        int ka = (kt + 1) * 32 + hi * 8;
                        afr[(kt + 1) & 1].h[0] = *(const v8bf*)(ap + ka);
                        afr[(kt + 1) & 1].h[1] = *(const v8bf*)(ap + ka + 16);
                    }
                    c = __builtin_amdgcn_wmma_f32_16x16x32_bf16(false, afr[kt & 1].v, false, bfr[kt].v, (short)0, c, false, false);
                }
                // pin schedule: [VMEM x4] then 6x[WMMA, VMEM x2] then [WMMA x2]
                __builtin_amdgcn_sched_group_barrier(SGB_VMEMRD, 4, 0);
#pragma unroll
                for (int g = 0; g < 6; ++g) {
                    __builtin_amdgcn_sched_group_barrier(SGB_WMMA, 1, 0);
                    __builtin_amdgcn_sched_group_barrier(SGB_VMEMRD, 2, 0);
                }
                __builtin_amdgcn_sched_group_barrier(SGB_WMMA, 2, 0);
            }
            const int rbase = mtile * 16 + hi * 8;
#pragma unroll
            for (int r = 0; r < 8; ++r) {
                int row = rbase + r;
                float v = c[r] + bias;
                v = (v > 0.f ? v : 0.f) + skip[(size_t)row * 256 + ncol];
                h2[(size_t)row * 256 + ncol] = v;
            }
        }
    } else {                                             // m-branch, K=128
        const int ncol = (ntile - 16) * 16 + nlo;
        V16U bfr[4];
        const unsigned short* bptr = m2T + (size_t)ncol * 128;
#pragma unroll
        for (int kt = 0; kt < 4; ++kt) {
            int kb = kt * 32 + hi * 8;
            bfr[kt].h[0] = *(const v8bf*)(bptr + kb);
            bfr[kt].h[1] = *(const v8bf*)(bptr + kb + 16);
        }
        const float bias = mb2[ncol];
        for (int mi = 0; mi < MLOOP; ++mi) {
            const int mtile = mbase + mi;
            const unsigned short* ap = m1 + (size_t)(mtile * 16 + nlo) * 128;
            v8f c = {};
            {
                V16U afr[2];
                afr[0].h[0] = *(const v8bf*)(ap + hi * 8);
                afr[0].h[1] = *(const v8bf*)(ap + hi * 8 + 16);
#pragma unroll
                for (int kt = 0; kt < 4; ++kt) {
                    if (kt + 1 < 4) {
                        int ka = (kt + 1) * 32 + hi * 8;
                        afr[(kt + 1) & 1].h[0] = *(const v8bf*)(ap + ka);
                        afr[(kt + 1) & 1].h[1] = *(const v8bf*)(ap + ka + 16);
                    }
                    c = __builtin_amdgcn_wmma_f32_16x16x32_bf16(false, afr[kt & 1].v, false, bfr[kt].v, (short)0, c, false, false);
                }
                __builtin_amdgcn_sched_group_barrier(SGB_VMEMRD, 4, 0);
#pragma unroll
                for (int g = 0; g < 2; ++g) {
                    __builtin_amdgcn_sched_group_barrier(SGB_WMMA, 1, 0);
                    __builtin_amdgcn_sched_group_barrier(SGB_VMEMRD, 2, 0);
                }
                __builtin_amdgcn_sched_group_barrier(SGB_WMMA, 2, 0);
            }
            const int rbase = mtile * 16 + hi * 8;
#pragma unroll
            for (int r = 0; r < 8; ++r) {
                int row = rbase + r;
                float v = c[r] + bias;
                m2o[(size_t)row * 64 + ncol] = (v > 0.f ? v : 0.f);
            }
        }
    }
}

// ---------------- stage 4: tiny output matvecs + sigmoid (wave per row) ---------
__global__ void final_kernel(const float* __restrict__ h2, const float* __restrict__ m2,
                             const float* __restrict__ Wo, const float* __restrict__ bo,
                             const float* __restrict__ M3, const float* __restrict__ mb3,
                             float* __restrict__ outDisp, float* __restrict__ outMat) {
    const int row = blockIdx.x * 8 + (threadIdx.x >> 5);
    const int lane = threadIdx.x & 31;
    float a0 = 0.f, a1 = 0.f, a2 = 0.f, am = 0.f;
#pragma unroll
    for (int j = 0; j < 8; ++j) {
        int h = lane + 32 * j;
        float v = h2[(size_t)row * 256 + h];
        a0 = fmaf(v, Wo[h * 3 + 0], a0);
        a1 = fmaf(v, Wo[h * 3 + 1], a1);
        a2 = fmaf(v, Wo[h * 3 + 2], a2);
    }
#pragma unroll
    for (int j = 0; j < 2; ++j) {
        int h = lane + 32 * j;
        am = fmaf(m2[(size_t)row * 64 + h], M3[h], am);
    }
    for (int off = 16; off > 0; off >>= 1) {
        a0 += __shfl_xor(a0, off, 32);
        a1 += __shfl_xor(a1, off, 32);
        a2 += __shfl_xor(a2, off, 32);
        am += __shfl_xor(am, off, 32);
    }
    if (lane == 0) {
        outDisp[(size_t)row * 3 + 0] = (a0 + bo[0]) * DISP_SCALE;
        outDisp[(size_t)row * 3 + 1] = (a1 + bo[1]) * DISP_SCALE;
        outDisp[(size_t)row * 3 + 2] = (a2 + bo[2]) * DISP_SCALE;
        float s = am + mb3[0];
        outMat[row] = 1.f / (1.f + __expf(-s));
    }
}

// ---------------- host launcher ----------------
extern "C" void kernel_launch(void* const* d_in, const int* in_sizes, int n_in,
                              void* d_out, int out_size, void* d_ws, size_t ws_size,
                              hipStream_t stream) {
    (void)in_sizes; (void)n_in; (void)out_size; (void)ws_size;
    const float* tmpl = (const float*)d_in[0];
    const float* surf = (const float*)d_in[1];
    const float* gf   = (const float*)d_in[2];
    const float* pf   = (const float*)d_in[3];
    const float* W1   = (const float*)d_in[4];
    const float* b1   = (const float*)d_in[5];
    const float* W2   = (const float*)d_in[6];
    const float* b2   = (const float*)d_in[7];
    const float* Ws   = (const float*)d_in[8];
    const float* bs   = (const float*)d_in[9];
    const float* Wo   = (const float*)d_in[10];
    const float* bo   = (const float*)d_in[11];
    const float* M1   = (const float*)d_in[12];
    const float* mb1  = (const float*)d_in[13];
    const float* M2   = (const float*)d_in[14];
    const float* mb2  = (const float*)d_in[15];
    const float* M3   = (const float*)d_in[16];
    const float* mb3  = (const float*)d_in[17];

    char* ws = (char*)d_ws;
    size_t off = 0;
    auto alloc = [&](size_t bytes) { char* p = ws + off; off = (off + bytes + 255) & ~(size_t)255; return p; };
    unsigned short* ni    = (unsigned short*)alloc((size_t)BT * NIP * 2);   // 18.9 MB
    unsigned short* wcatT = (unsigned short*)alloc((size_t)NC1 * NIP * 2);  // 0.37 MB
    float*          gbias = (float*)        alloc((size_t)BB * NC1 * 4);
    unsigned short* w2T   = (unsigned short*)alloc((size_t)256 * 256 * 2);
    unsigned short* m2T   = (unsigned short*)alloc((size_t)64 * 128 * 2);
    unsigned short* h1    = (unsigned short*)alloc((size_t)BT * 256 * 2);   // 16.8 MB
    float*          skip  = (float*)        alloc((size_t)BT * 256 * 4);   // 33.6 MB
    unsigned short* m1    = (unsigned short*)alloc((size_t)BT * 128 * 2);   // 8.4 MB
    float*          h2    = (float*)        alloc((size_t)BT * 256 * 4);   // 33.6 MB
    float*          m2    = (float*)        alloc((size_t)BT * 64 * 4);    // 8.4 MB

    float* outDisp = (float*)d_out;               // [B,T,3]
    float* outMat  = outDisp + (size_t)BT * 3;    // [B,T]

    // weight prep
    prep_wcat_kernel<<<(NC1 * NIP + 255) / 256, 256, 0, stream>>>(W1, Ws, M1, wcatT);
    prep_w2m2_kernel<<<(256 * 256 + 64 * 128 + 255) / 256, 256, 0, stream>>>(W2, M2, w2T, m2T);
    prep_gbias_kernel<<<(BB * NC1 + 255) / 256, 256, 0, stream>>>(gf, W1, b1, Ws, bs, M1, mb1, gbias);

    // stage 1: local features → bf16 ni
    local_feat_kernel<<<BT / 8, 256, 0, stream>>>(tmpl, surf, pf, ni);

    // stage 2: fused GEMM1 (640 cols), B in registers, async double-buffered A
    gemm1_kernel<<<dim3(BT / 16 / MLOOP, NC1 / (16 * 8)), 256, 0, stream>>>(ni, wcatT, gbias, h1, skip, m1);

    // stage 3: fused GEMM2 (h-branch 16 N-tiles + m-branch 4 N-tiles)
    gemm2_kernel<<<(BT / 16 / MLOOP) * 20 / 8, 256, 0, stream>>>(h1, w2T, b2, skip, m1, m2T, mb2, h2, m2);

    // stage 4: output heads
    final_kernel<<<BT / 8, 256, 0, stream>>>(h2, m2, Wo, bo, M3, mb3, outDisp, outMat);
}
